// WindowAttention_13795434955392
// MI455X (gfx1250) — matrix operand order repaired and checked by reading, not compiled, not run on hardware
//
#include <hip/hip_runtime.h>
#include <hip/hip_bf16.h>

typedef __attribute__((ext_vector_type(16))) _Float16 v16h;
typedef __attribute__((ext_vector_type(8)))  float    v8f;

// ---------------------------------------------------------------------------
// CDNA5 WMMA 16x16x32 f16 fragment layout (wave32).
// A (16x32, MxK): lane L: m = L&15, half = L>>4.
//   VGPR v in 0..7 holds K = 2v + 16*half (pair): this file's contiguous order, not the ISA table's.
// B (32x16, KxN): lane L: n = L&15, half = L>>4; element e holds K = e + 16*half (same contiguous order as A, chosen here).
// C/D (16x16 f32): lane L: col n = L&15; VGPR v holds row m = v + 8*(L>>4).
// ---------------------------------------------------------------------------
__device__ __forceinline__ int a_kbase(int v, int half) {
  return 2 * v + 16 * half;
}

// ---------------------------------------------------------------------------
// Generic GEMM: out[m,n] = sum_k A[m,k] * Bt[n,k] + bias[n]
// A: (M,K) f32 row-major, Bt: (N,K) f32 row-major, out: (M,N) f32.
// K is a compile-time constant (192 here) -> fully unrolled WMMA pipeline.
// Block = 128 threads = 4 waves; wave w computes tile (blockIdx.x*4+w, blockIdx.y).
// ---------------------------------------------------------------------------
template <int K>
__global__ __launch_bounds__(128)
void gemm_bias_f32_wmma(const float* __restrict__ A, const float* __restrict__ Bt,
                        const float* __restrict__ bias, float* __restrict__ out,
                        int M, int N) {
  const int lane = threadIdx.x & 31;
  const int wave = threadIdx.x >> 5;
  const int half = lane >> 4;
  const int l16  = lane & 15;
  const int tileM = blockIdx.x * 4 + wave;
  const int tileN = blockIdx.y;
  if (tileM * 16 >= M) return;

  const int m = tileM * 16 + l16;   // A-frag row for this lane
  const int n = tileN * 16 + l16;   // B-frag / C col for this lane

  const float* arow_base = A  + (size_t)m * K;
  const float* brow_base = Bt + (size_t)n * K;

  v8f acc = {};
#pragma unroll
  for (int k0 = 0; k0 < K; k0 += 32) {
    // A fragment: 16 contiguous floats per lane -> 4x b128 loads
    float fa[16];
    const float* arow = arow_base + k0;
    *(float4*)&fa[0]  = ((const float4*)(arow + 16 * half))[0];
    *(float4*)&fa[4]  = ((const float4*)(arow + 16 * half))[1];
    *(float4*)&fa[8]  = ((const float4*)(arow + 16 * half))[2];
    *(float4*)&fa[12] = ((const float4*)(arow + 16 * half))[3];
    // B fragment: 16 contiguous floats per lane -> 4x b128 loads
    float fb[16];
    const float* brow = brow_base + k0 + 16 * half;
    *(float4*)&fb[0]  = ((const float4*)brow)[0];
    *(float4*)&fb[4]  = ((const float4*)brow)[1];
    *(float4*)&fb[8]  = ((const float4*)brow)[2];
    *(float4*)&fb[12] = ((const float4*)brow)[3];

    v16h a, b;
#pragma unroll
    for (int v = 0; v < 8; ++v) {
      int fi = (v < 4) ? 2 * v : 8 + 2 * (v - 4);
      a[2 * v]     = (_Float16)fa[fi];
      a[2 * v + 1] = (_Float16)fa[fi + 1];
    }
#pragma unroll
    for (int e = 0; e < 16; ++e) b[e] = (_Float16)fb[e];
    acc = __builtin_amdgcn_wmma_f32_16x16x32_f16(false, a, false, b,
                                                 (short)0, acc, false, false);
  }
  const float bv = bias[n];
#pragma unroll
  for (int v = 0; v < 8; ++v) {
    int row = tileM * 16 + v + 8 * half;
    out[(size_t)row * N + n] = acc[v] + bv;
  }
}

// ---------------------------------------------------------------------------
// Focal window attention: one block per (window, head).
//   q/k/v fine:  qkvf (100352, 576)  cols [0:192)=q [192:384)=k [384:576)=v
//   k/v pooled:  qkvp (2048, 576)
//   keys: 49 window + 132 rolled-ring + 49 pooled = 230, padded to 240.
// ---------------------------------------------------------------------------
#define NKEY  230
#define NKPAD 240
#define SROWS 50                         // 49 query rows + 1 dump row
#define SCALE 0.17677669529663689f
#define POOLED_FLAG (1 << 28)

__global__ __launch_bounds__(128)
void focal_attn_wmma(const float* __restrict__ qkvf, const float* __restrict__ qkvp,
                     const float* __restrict__ rpbT,   // (169, 6)
                     const float* __restrict__ rpbN,   // (6, 49, 132)
                     const float* __restrict__ rpbW,   // (6, 169)
                     float* __restrict__ aout) {       // (100352, 192)
  __shared__ float    S[SROWS * NKPAD];   // scores -> probabilities (f32); row 49 = dump
  __shared__ _Float16 KV[NKPAD * 32];     // staged K, then reused for V
  __shared__ int      keytok[NKPAD];      // key -> source token (|POOLED_FLAG), -1 = zero
  __shared__ int      validg[132];        // VALID_ROLLED table

  const int winId = blockIdx.x;           // 0..2047
  const int head  = blockIdx.y;           // 0..5
  const int b  = winId >> 8;
  const int w  = winId & 255;
  const int wr = w >> 4, wc = w & 15;
  const int tid  = threadIdx.x;
  const int lane = tid & 31, wave = tid >> 5;
  const int half = lane >> 4, l16 = lane & 15;

  // ---- VALID_ROLLED: sorted g in 0..195 passing the per-shift ring mask ----
  if (tid == 0) {
    int cnt = 0;
    for (int g = 0; g < 196; ++g) {
      int s = g / 49, t = g % 49, r = t / 7, c = t % 7;
      bool ok = (s == 0) ? (r >= 4 || c >= 4)
              : (s == 1) ? (r >= 4 || c <  3)
              : (s == 2) ? (r <  3 || c >= 4)
              :            (r <  3 || c <  3);
      if (ok) validg[cnt++] = g;
    }
  }
  __syncthreads();

  // ---- key gather table ----
  for (int j = tid; j < NKPAD; j += blockDim.x) {
    int tok = -1;
    if (j < 49) {                                    // in-window keys
      int r = j / 7, c = j % 7;
      tok = (b * 112 + wr * 7 + r) * 112 + (wc * 7 + c);
    } else if (j < 181) {                            // rolled neighbor ring
      int g = validg[j - 49];
      int s = g / 49, t = g % 49, r = t / 7, c = t % 7;
      int sr = (s < 2) ? -3 : 3;
      int sc = (s & 1) ?  3 : -3;
      int row = (wr * 7 + r - sr + 112) % 112;       // roll(shift s): src = i - s mod H
      int col = (wc * 7 + c - sc + 112) % 112;
      tok = (b * 112 + row) * 112 + col;
    } else if (j < NKEY) {                           // pooled unfold, pad 3
      int t = j - 181, pr = t / 7, pc = t % 7;
      int prow = wr + pr - 3, pcol = wc + pc - 3;
      if (prow >= 0 && prow < 16 && pcol >= 0 && pcol < 16)
        tok = (((b * 16 + prow) * 16 + pcol)) | POOLED_FLAG;
    }
    keytok[j] = tok;
  }
  __syncthreads();

  // ---- stage K (chanBase=192) / V (chanBase=384) into KV as f16 ----
  // One wave iteration covers all 32 channels of one key -> coalesced 128B,
  // and the fine/pooled branch is wave-uniform.
  auto stage = [&](int chanBase) {
    for (int idx = tid; idx < NKPAD * 32; idx += blockDim.x) {
      int j = idx >> 5, c = idx & 31;
      int tok = keytok[j];
      float val = 0.f;
      if (tok >= 0) {
        if (tok & POOLED_FLAG)
          val = qkvp[(size_t)(tok & (POOLED_FLAG - 1)) * 576 + chanBase + head * 32 + c];
        else
          val = qkvf[(size_t)tok * 576 + chanBase + head * 32 + c];
      }
      KV[j * 32 + c] = (_Float16)val;
    }
  };
  stage(192);
  __syncthreads();

  // ---- pass 1: S = (Q*scale) @ K^T (raw scores) ; wave owns M-tile `wave` ----
  v16h aq;
  {
    int m = wave * 16 + l16;
    if (m < 49) {
      int tok = (b * 112 + wr * 7 + m / 7) * 112 + (wc * 7 + m % 7);
      const float* qrow = qkvf + (size_t)tok * 576 + head * 32;
#pragma unroll
      for (int v = 0; v < 8; ++v) {
        int kb = a_kbase(v, half);
        aq[2 * v]     = (_Float16)(qrow[kb]     * SCALE);
        aq[2 * v + 1] = (_Float16)(qrow[kb + 1] * SCALE);
      }
    } else {
#pragma unroll
      for (int e = 0; e < 16; ++e) aq[e] = (_Float16)0.f;
    }
  }

  for (int nt = 0; nt < 15; ++nt) {
    v16h bk;
    {
      const _Float16* kr = &KV[(nt * 16 + l16) * 32 + 16 * half];
#pragma unroll
      for (int e = 0; e < 16; ++e) bk[e] = kr[e];
    }
    v8f s = {};
    s = __builtin_amdgcn_wmma_f32_16x16x32_f16(false, aq, false, bk,
                                               (short)0, s, false, false);
    const int key = nt * 16 + l16;
    // Branchless store: invalid rows (only wave 3 padding lanes) are clamped
    // onto dump row 49, which is never read afterwards.
#pragma unroll
    for (int v = 0; v < 8; ++v) {
      int row = wave * 16 + v + 8 * half;
      int srow = (row < 49) ? row : 49;
      S[srow * NKPAD + key] = s[v];     // raw score; bias added below
    }
  }
  __syncthreads();

  // ---- cooperative bias + pooled-OOB mask add (flat, near-uniform flow) ----
  for (int idx = tid; idx < 49 * NKEY; idx += blockDim.x) {
    int row = idx / NKEY;
    int key = idx - row * NKEY;
    int qr = row / 7, qc = row - 7 * qr;
    float bias;
    if (key < 49) {
      int kr = key / 7, kc = key - 7 * kr;
      bias = rpbT[((qr - kr + 6) * 13 + (qc - kc + 6)) * 6 + head];
    } else if (key < 181) {
      bias = rpbN[((size_t)head * 49 + row) * 132 + (key - 49)];
    } else {
      int t = key - 181;
      int kr = t / 7, kc = t - 7 * kr;
      bias = rpbW[head * 169 + (qr - kr + 6) * 13 + (qc - kc + 6)];
      if (keytok[key] < 0) bias -= 100.f;          // pooled zero-pad mask
    }
    S[row * NKPAD + key] += bias;
  }
  __syncthreads();

  // ---- softmax over 230 keys, fp32, one thread per query row ----
  if (tid < 49) {
    float* srow = &S[tid * NKPAD];
    float mx = -1e30f;
    for (int j = 0; j < NKEY; ++j) mx = fmaxf(mx, srow[j]);
    float sum = 0.f;
    for (int j = 0; j < NKEY; ++j) { float e = __expf(srow[j] - mx); srow[j] = e; sum += e; }
    float inv = 1.f / sum;
    for (int j = 0; j < NKEY; ++j) srow[j] *= inv;
    for (int j = NKEY; j < NKPAD; ++j) srow[j] = 0.f;   // pad cols -> P = 0
  }
  __syncthreads();

  stage(384);   // V replaces K in the same LDS buffer
  __syncthreads();

  // ---- pass 2: O = P @ V ; wave owns M-tile `wave`, loops 2 N-tiles ----
  const int m2 = wave * 16 + l16;
  const float* prow = &S[(m2 < 49 ? m2 : 0) * NKPAD];
#pragma unroll
  for (int nt = 0; nt < 2; ++nt) {
    v8f acc = {};
#pragma unroll
    for (int kk = 0; kk < 8; ++kk) {
      v16h ap, bv;
#pragma unroll
      for (int v = 0; v < 8; ++v) {
        int kb = kk * 32 + a_kbase(v, half);
        float p0 = (m2 < 49 && kb     < NKPAD) ? prow[kb]     : 0.f;
        float p1 = (m2 < 49 && kb + 1 < NKPAD) ? prow[kb + 1] : 0.f;
        ap[2 * v]     = (_Float16)p0;
        ap[2 * v + 1] = (_Float16)p1;
      }
#pragma unroll
      for (int e = 0; e < 16; ++e) {
        int krow = kk * 32 + 16 * half + e;
        bv[e] = (krow < NKPAD) ? KV[krow * 32 + nt * 16 + l16] : (_Float16)0.f;
      }
      acc = __builtin_amdgcn_wmma_f32_16x16x32_f16(false, ap, false, bv,
                                                   (short)0, acc, false, false);
    }
#pragma unroll
    for (int v = 0; v < 8; ++v) {
      int row = wave * 16 + v + 8 * half;
      if (row < 49) {
        size_t o = ((size_t)winId * 49 + row) * 192 + head * 32 + nt * 16 + l16;
        aout[o] = acc[v];
      }
    }
  }
}

// ---------------------------------------------------------------------------
extern "C" void kernel_launch(void* const* d_in, const int* in_sizes, int n_in,
                              void* d_out, int out_size, void* d_ws, size_t ws_size,
                              hipStream_t stream) {
  const float* x        = (const float*)d_in[0];   // (8,112,112,192)
  const float* x_pooled = (const float*)d_in[1];   // (8,16,16,192)
  const float* qkv_w    = (const float*)d_in[2];   // (576,192)
  const float* qkv_b    = (const float*)d_in[3];   // (576)
  const float* proj_w   = (const float*)d_in[4];   // (192,192)
  const float* proj_b   = (const float*)d_in[5];   // (192)
  const float* rpbT     = (const float*)d_in[6];   // (169,6)
  const float* rpbN     = (const float*)d_in[7];   // (1,6,49,132)
  const float* rpbW     = (const float*)d_in[8];   // (6,169)
  float* out = (float*)d_out;                      // (100352,192)

  const int M_FINE = 8 * 112 * 112;   // 100352
  const int M_POOL = 8 * 16 * 16;     // 2048

  float* qkvf = (float*)d_ws;                            // (100352,576)
  float* qkvp = qkvf + (size_t)M_FINE * 576;             // (2048,576)
  float* aout = qkvp + (size_t)M_POOL * 576;             // (100352,192)

  // fine QKV:   (100352,192) x (576,192)^T
  gemm_bias_f32_wmma<192><<<dim3(M_FINE / 64, 576 / 16), 128, 0, stream>>>(
      x, qkv_w, qkv_b, qkvf, M_FINE, 576);
  // pooled QKV: (2048,192) x (576,192)^T
  gemm_bias_f32_wmma<192><<<dim3(M_POOL / 64, 576 / 16), 128, 0, stream>>>(
      x_pooled, qkv_w, qkv_b, qkvp, M_POOL, 576);
  // attention: 2048 windows x 6 heads
  focal_attn_wmma<<<dim3(2048, 6), 128, 0, stream>>>(
      qkvf, qkvp, rpbT, rpbN, rpbW, aout);
  // projection: (100352,192) x (192,192)^T
  gemm_bias_f32_wmma<192><<<dim3(M_FINE / 64, 192 / 16), 128, 0, stream>>>(
      aout, proj_w, proj_b, out, M_FINE, 192);
}
